// MambaSSM_17617955848828
// MI455X (gfx1250) — matrix operand order repaired and checked by reading, not compiled, not run on hardware
//
#include <hip/hip_runtime.h>
#include <hip/hip_bf16.h>
#include <math.h>
#include <stdint.h>

// ---------------------------------------------------------------------------
// Mamba-2 SSD for MI455X (gfx1250, wave32, WMMA bf16).
// Big GEMMs: bf16 WMMA with ASYNCcnt-tracked async-to-LDS weight staging
// (double buffered). Scalar decay chain (softplus/cumsum/exp/scan) stays f32.
// ---------------------------------------------------------------------------

typedef __bf16 bf16;
typedef __attribute__((ext_vector_type(16))) __bf16 v16bf;
typedef __attribute__((ext_vector_type(8)))  __bf16 v8bf;
typedef __attribute__((ext_vector_type(8)))  float  v8f;

#define BATCH     2
#define SEQLEN    4096
#define DMODEL    2048
#define INNER     4096
#define NHEADS    64
#define HEADDIM   64
#define DSTATE    128
#define CHUNK     64
#define NCHUNK    64                 // per batch
#define KIN       4416               // inner + 2*n + h
#define MROWS     (BATCH * SEQLEN)   // 8192

__device__ __forceinline__ v8f wmma_bf16(v16bf a, v16bf b, v8f c) {
  return __builtin_amdgcn_wmma_f32_16x16x32_bf16(false, a, false, b, (short)0, c,
                                                 false, false);
}

// A-fragment (16x32, m rows, k cols) from row-major [m][k] memory.
// Lanes 0-15 hold K {0..7,16..23}, lanes 16-31 hold K {8..15,24..31}.
__device__ __forceinline__ v16bf load_a_frag(const bf16* __restrict__ base,
                                             int ldk, int m0, int k0) {
  int lane = threadIdx.x & 31;
  int row  = m0 + (lane & 15);
  int kb   = k0 + ((lane >> 4) << 3);
  const bf16* p = base + (size_t)row * ldk + kb;
  v8bf lo = *(const v8bf*)p;
  v8bf hi = *(const v8bf*)(p + 16);
  v16bf r;
#pragma unroll
  for (int i = 0; i < 8; ++i) { r[i] = lo[i]; r[i + 8] = hi[i]; }
  return r;
}

// B-fragment (32x16) from row-major [col][k] memory (i.e. B^T); col=lane%16,
// lanes 0-15 K {0..7,16..23}, lanes 16-31 K {8..15,24..31}.
__device__ __forceinline__ v16bf load_b_frag(const bf16* __restrict__ base,
                                             int ldk, int n0, int k0) {
  int lane = threadIdx.x & 31;
  int col  = n0 + (lane & 15);
  int kb   = k0 + ((lane >> 4) << 3);
  const bf16* p = base + (size_t)col * ldk + kb;
  v8bf lo = *(const v8bf*)p;
  v8bf hi = *(const v8bf*)(p + 16);
  v16bf r;
#pragma unroll
  for (int i = 0; i < 8; ++i) { r[i] = lo[i]; r[i + 8] = hi[i]; }
  return r;
}

// gfx1250 async global->LDS copy (ASYNCcnt-tracked); lds_off is a workgroup LDS byte address.
__device__ __forceinline__ void async_load_b128(unsigned lds_off,
                                                const bf16* gaddr) {
  asm volatile("global_load_async_to_lds_b128 %0, %1, off"
               :: "v"(lds_off), "v"(gaddr)
               : "memory");
}

// ---------------------------------------------------------------------------
// K0: f32 -> bf16 pack
__global__ void pack_bf16_kernel(const float* __restrict__ src,
                                 bf16* __restrict__ dst, long n) {
  long i = (long)blockIdx.x * blockDim.x + threadIdx.x;
  long stride = (long)gridDim.x * blockDim.x;
  for (; i < n; i += stride) dst[i] = (bf16)src[i];
}

// ---------------------------------------------------------------------------
// K1/K8: NT GEMM, C[M,N] f32 = A[M,K]bf16 * B[N,K]bf16^T.
// Block tile 128M x 64N, 8 waves of 32x32; B tile (64x32, 4KB) staged in LDS
// via global_load_async_to_lds_b128, double buffered on ASYNCcnt.
__global__ __launch_bounds__(256) void gemm_nt_kernel(
    const bf16* __restrict__ A, const bf16* __restrict__ B,
    float* __restrict__ C, int M, int N, int K) {
  __shared__ __align__(16) bf16 tileB[2][64 * 32];   // [buf][col][k]
  int tid  = threadIdx.x;
  int w    = tid >> 5;
  int lane = tid & 31;
  int m0   = blockIdx.y * 128 + (w >> 1) * 32;
  int nblk = blockIdx.x * 64;
  int nw   = (w & 1) * 32;

  // per-thread async-copy slot: one b128 of the 64x32 tile
  unsigned ldsbase = (unsigned)(uintptr_t)(&tileB[0][0]);
  int colL = tid & 63, seg = tid >> 6;               // seg 0..3 (8 elems)
  unsigned loff0 = ldsbase + (unsigned)(colL * 64 + seg * 16);
  const bf16* gsrc = B + (size_t)(nblk + colL) * K + seg * 8;

  // prologue: K-step 0 -> buffer 0
  async_load_b128(loff0, gsrc);

  v8f acc[2][2] = {};
  int nIter = K / 32;
  for (int it = 0; it < nIter; ++it) {
    int cur = it & 1;
    if (it + 1 < nIter) {
      async_load_b128(loff0 + (unsigned)((1 - cur) * 4096),
                      gsrc + (size_t)(it + 1) * 32);
      asm volatile("s_wait_asynccnt 0x1" ::: "memory");  // current buf done
    } else {
      asm volatile("s_wait_asynccnt 0x0" ::: "memory");
    }
    __syncthreads();                                     // all waves' copies
    int k0 = it * 32;
    v16bf a0 = load_a_frag(A, K, m0, k0);
    v16bf a1 = load_a_frag(A, K, m0 + 16, k0);
    int colb = nw + (lane & 15);
    int kb   = (lane >> 4) << 3;
    const bf16* q0 = &tileB[cur][colb * 32 + kb];
    const bf16* q1 = &tileB[cur][(colb + 16) * 32 + kb];
    v8bf b0lo = *(const v8bf*)q0, b0hi = *(const v8bf*)(q0 + 16);
    v8bf b1lo = *(const v8bf*)q1, b1hi = *(const v8bf*)(q1 + 16);
    v16bf b0, b1;
#pragma unroll
    for (int i = 0; i < 8; ++i) {
      b0[i] = b0lo[i]; b0[i + 8] = b0hi[i];
      b1[i] = b1lo[i]; b1[i + 8] = b1hi[i];
    }
    acc[0][0] = wmma_bf16(a0, b0, acc[0][0]);
    acc[0][1] = wmma_bf16(a0, b1, acc[0][1]);
    acc[1][0] = wmma_bf16(a1, b0, acc[1][0]);
    acc[1][1] = wmma_bf16(a1, b1, acc[1][1]);
    __syncthreads();                                     // safe to overwrite
  }
  int tl = lane & 15, th = lane >> 4;
#pragma unroll
  for (int mt = 0; mt < 2; ++mt)
#pragma unroll
    for (int nt = 0; nt < 2; ++nt)
#pragma unroll
      for (int v = 0; v < 8; ++v) {
        int row = m0 + mt * 16 + v + th * 8;
        C[(size_t)row * N + nblk + nw + nt * 16 + tl] = acc[mt][nt][v];
      }
}

// ---------------------------------------------------------------------------
// K2: per-row prep. dt=softplus(dt_raw+bias); dtA=-exp(A_log)*dt;
// xdtT[b,c,h,p,s]=x*dt (bf16); Bc/Cc[b,l,n]; BT[b,c,n,s].
__global__ __launch_bounds__(256) void prep_kernel(
    const float* __restrict__ xbcdt, const float* __restrict__ A_log,
    const float* __restrict__ dt_bias, bf16* __restrict__ xdtT,
    bf16* __restrict__ Bc, bf16* __restrict__ Cc, bf16* __restrict__ BT,
    float* __restrict__ dtA) {
  __shared__ float sdt[NHEADS];
  int r   = blockIdx.x;                 // b*4096 + l
  int tid = threadIdx.x;
  const float* row = xbcdt + (size_t)r * KIN;
  if (tid < NHEADS) {
    float raw = row[INNER + 2 * DSTATE + tid] + dt_bias[tid];
    float dt  = raw > 20.f ? raw : log1pf(expf(raw));   // softplus
    sdt[tid]  = dt;
    dtA[(size_t)r * NHEADS + tid] = -expf(A_log[tid]) * dt;
  }
  __syncthreads();
  int l = r & (SEQLEN - 1);
  int ci = l >> 6, s = l & 63;
  size_t chunk = (size_t)((r >> 12) * NCHUNK + ci);     // b*64+ci
  for (int col = tid; col < INNER; col += 256) {
    int hh = col >> 6, pp = col & 63;
    float v = row[col] * sdt[hh];
    xdtT[((chunk * NHEADS + hh) * HEADDIM + pp) * CHUNK + s] = (bf16)v;
  }
  if (tid < DSTATE) {
    float v = row[INNER + tid];
    Bc[(size_t)r * DSTATE + tid] = (bf16)v;
    BT[(chunk * DSTATE + tid) * CHUNK + s] = (bf16)v;
  } else {
    float v = row[INNER + DSTATE + (tid - DSTATE)];
    Cc[(size_t)r * DSTATE + (tid - DSTATE)] = (bf16)v;
  }
}

// ---------------------------------------------------------------------------
// K3: intra-chunk inclusive cumsum of dtA.  Acs[b,c,h,t], Aend[b,h,c].
__global__ __launch_bounds__(64) void cumsum_kernel(
    const float* __restrict__ dtA, float* __restrict__ Acs,
    float* __restrict__ Aend) {
  int bc = blockIdx.x;                  // b*64+ci
  int hh = threadIdx.x;
  size_t rbase = (size_t)bc * CHUNK;    // first seq row of chunk
  size_t obase = ((size_t)bc * NHEADS + hh) * CHUNK;
  float acc = 0.f;
  for (int t = 0; t < CHUNK; ++t) {
    acc += dtA[(rbase + t) * NHEADS + hh];
    Acs[obase + t] = acc;
  }
  int bb = bc >> 6, ci = bc & 63;
  Aend[((size_t)bb * NHEADS + hh) * NCHUNK + ci] = acc;
}

// ---------------------------------------------------------------------------
// K4: G[t,s] = C[t,:]·B[s,:]  (per chunk, WMMA NT, shared across heads).
__global__ __launch_bounds__(128) void gmat_kernel(
    const bf16* __restrict__ Cc, const bf16* __restrict__ Bc,
    bf16* __restrict__ G) {
  int bc = blockIdx.x;
  int w = threadIdx.x >> 5, lane = threadIdx.x & 31;
  const bf16* Cb = Cc + (size_t)bc * CHUNK * DSTATE;
  const bf16* Bb = Bc + (size_t)bc * CHUNK * DSTATE;
  int t0 = w * 16;
  v8f acc[4] = {};
  for (int k0 = 0; k0 < DSTATE; k0 += 32) {
    v16bf a = load_a_frag(Cb, DSTATE, t0, k0);
#pragma unroll
    for (int nt = 0; nt < 4; ++nt) {
      v16bf b = load_b_frag(Bb, DSTATE, nt * 16, k0);
      acc[nt] = wmma_bf16(a, b, acc[nt]);
    }
  }
  int tl = lane & 15, th = lane >> 4;
  bf16* Gb = G + (size_t)bc * CHUNK * CHUNK;
#pragma unroll
  for (int nt = 0; nt < 4; ++nt)
#pragma unroll
    for (int v = 0; v < 8; ++v)
      Gb[(size_t)(t0 + v + th * 8) * CHUNK + nt * 16 + tl] = (bf16)acc[nt][v];
}

// ---------------------------------------------------------------------------
// K5: local chunk states St[b,c,h,p,n] = sum_s xdt[s,p]*decay[s]*B[s,n]. (bf16)
__global__ __launch_bounds__(128) void states_kernel(
    const bf16* __restrict__ xdtT, const bf16* __restrict__ BT,
    const float* __restrict__ Acs, bf16* __restrict__ St) {
  int bch = blockIdx.x;                 // (b*64+ci)*64+hh
  int bc  = bch >> 6;
  __shared__ float sdec[CHUNK];
  int tid = threadIdx.x;
  const float* cs = Acs + (size_t)bch * CHUNK;
  if (tid < CHUNK) sdec[tid] = __expf(cs[CHUNK - 1] - cs[tid]);
  __syncthreads();
  int w = tid >> 5, lane = tid & 31;
  int m0 = w * 16;                                      // p tile
  const bf16* xT = xdtT + (size_t)bch * HEADDIM * CHUNK; // [p][s]
  const bf16* Bt = BT + (size_t)bc * DSTATE * CHUNK;     // [n][s]
  v8f acc[8] = {};
#pragma unroll
  for (int k0 = 0; k0 < CHUNK; k0 += 32) {
    int kb = k0 + ((lane >> 4) << 3);
    const bf16* p0 = xT + (size_t)(m0 + (lane & 15)) * CHUNK + kb;
    v8bf lo = *(const v8bf*)p0;
    v8bf hi = *(const v8bf*)(p0 + 16);
    v16bf a;
#pragma unroll
    for (int j = 0; j < 8; ++j) {
      a[j]     = (bf16)((float)lo[j] * sdec[kb + j]);
      a[j + 8] = (bf16)((float)hi[j] * sdec[kb + 16 + j]);
    }
#pragma unroll
    for (int nt = 0; nt < 8; ++nt) {
      v16bf b = load_b_frag(Bt, CHUNK, nt * 16, k0);
      acc[nt] = wmma_bf16(a, b, acc[nt]);
    }
  }
  int tl = lane & 15, th = lane >> 4;
  bf16* out = St + (size_t)bch * HEADDIM * DSTATE;       // [p][n]
#pragma unroll
  for (int nt = 0; nt < 8; ++nt)
#pragma unroll
    for (int v = 0; v < 8; ++v)
      out[(size_t)(m0 + v + th * 8) * DSTATE + nt * 16 + tl] = (bf16)acc[nt][v];
}

// ---------------------------------------------------------------------------
// K6: sequential chunk-level scan per (b,h). Overwrites St[ci] in place with
// the prefix (incoming) state; writes final_state f32 to d_out tail.
__global__ __launch_bounds__(256) void scan_kernel(
    const float* __restrict__ Aend, bf16* __restrict__ St,
    float* __restrict__ final_state) {
  int bh = blockIdx.x;                  // b*64+hh
  int bb = bh >> 6, hh = bh & 63;
  int tid = threadIdx.x;
  float S[32];
#pragma unroll
  for (int j = 0; j < 32; ++j) S[j] = 0.f;
  const float* ae = Aend + (size_t)bh * NCHUNK;
  for (int ci = 0; ci < NCHUNK; ++ci) {
    float dec = __expf(ae[ci]);
    bf16* base = St + ((size_t)(bb * NCHUNK + ci) * NHEADS + hh) * (HEADDIM * DSTATE);
    if (ci + 1 < NCHUNK) {               // prefetch next chunk's tile
      const bf16* nxt = base + (size_t)NHEADS * HEADDIM * DSTATE;
      __builtin_prefetch(nxt + tid * 32, 0, 1);
    }
#pragma unroll
    for (int j = 0; j < 32; ++j) {
      int e = j * 256 + tid;            // p*128+n
      float local = (float)base[e];
      base[e] = (bf16)S[j];             // prefix state for this chunk
      S[j] = S[j] * dec + local;
    }
  }
  float* fs = final_state + (size_t)bh * (HEADDIM * DSTATE); // (b,h,p,n)
#pragma unroll
  for (int j = 0; j < 32; ++j) fs[j * 256 + tid] = S[j];
}

// ---------------------------------------------------------------------------
// K7: Y[t,p] = sum_s G[t,s]*L[t,s]*xdt[s,p] + exp(cs[t])*sum_n C[t,n]*Sin[p,n]
__global__ __launch_bounds__(128) void y_kernel(
    const bf16* __restrict__ G, const bf16* __restrict__ Cc,
    const bf16* __restrict__ xdtT, const bf16* __restrict__ St,
    const float* __restrict__ Acs, bf16* __restrict__ Yb) {
  int bch = blockIdx.x;
  int bc = bch >> 6, hh = bch & 63;
  __shared__ float scs[CHUNK];
  int tid = threadIdx.x;
  const float* cs = Acs + (size_t)bch * CHUNK;
  if (tid < CHUNK) scs[tid] = cs[tid];
  __syncthreads();
  int w = tid >> 5, lane = tid & 31;
  int t0 = w * 16;
  int trow = t0 + (lane & 15);
  float cst = scs[trow];
  const bf16* Gb = G + (size_t)bc * CHUNK * CHUNK;        // [t][s]
  const bf16* Cb = Cc + (size_t)bc * CHUNK * DSTATE;      // [t][n]
  const bf16* xT = xdtT + (size_t)bch * HEADDIM * CHUNK;  // [p][s]
  const bf16* Sb = St + (size_t)bch * HEADDIM * DSTATE;   // [p][n] prefix
  v8f acc[4] = {};
  // ---- Y_diag: A = G .* L (masked exp of cumsum differences)
#pragma unroll
  for (int k0 = 0; k0 < CHUNK; k0 += 32) {
    int kb = k0 + ((lane >> 4) << 3);
    const bf16* gp = Gb + (size_t)trow * CHUNK + kb;
    v8bf glo = *(const v8bf*)gp;
    v8bf ghi = *(const v8bf*)(gp + 16);
    v16bf a;
#pragma unroll
    for (int j = 0; j < 8; ++j) {
      int s1 = kb + j, s2 = kb + 16 + j;
      float l1 = (s1 <= trow) ? __expf(cst - scs[s1]) : 0.f;
      float l2 = (s2 <= trow) ? __expf(cst - scs[s2]) : 0.f;
      a[j]     = (bf16)((float)glo[j] * l1);
      a[j + 8] = (bf16)((float)ghi[j] * l2);
    }
#pragma unroll
    for (int nt = 0; nt < 4; ++nt) {
      v16bf b = load_b_frag(xT, CHUNK, nt * 16, k0);
      acc[nt] = wmma_bf16(a, b, acc[nt]);
    }
  }
  // ---- Y_off: A = C * exp(cs[t]) ; B = prefix states [p][n]
  float dto = __expf(cst);
#pragma unroll
  for (int k0 = 0; k0 < DSTATE; k0 += 32) {
    int kb = k0 + ((lane >> 4) << 3);
    const bf16* cp = Cb + (size_t)trow * DSTATE + kb;
    v8bf clo = *(const v8bf*)cp;
    v8bf chi = *(const v8bf*)(cp + 16);
    v16bf a;
#pragma unroll
    for (int j = 0; j < 8; ++j) {
      a[j]     = (bf16)((float)clo[j] * dto);
      a[j + 8] = (bf16)((float)chi[j] * dto);
    }
#pragma unroll
    for (int nt = 0; nt < 4; ++nt) {
      v16bf b = load_b_frag(Sb, DSTATE, nt * 16, k0);
      acc[nt] = wmma_bf16(a, b, acc[nt]);
    }
  }
  // store Ybuf row-major (b*l) x (h*p), bf16, ready as GEMM2 A-operand
  int tl = lane & 15, th = lane >> 4;
  size_t lbase = (size_t)bc * CHUNK;    // = b*4096 + ci*64
#pragma unroll
  for (int nt = 0; nt < 4; ++nt)
#pragma unroll
    for (int v = 0; v < 8; ++v) {
      int t = t0 + v + th * 8;
      int p = nt * 16 + tl;
      Yb[(lbase + t) * INNER + hh * HEADDIM + p] = (bf16)acc[nt][v];
    }
}

// ---------------------------------------------------------------------------
extern "C" void kernel_launch(void* const* d_in, const int* in_sizes, int n_in,
                              void* d_out, int out_size, void* d_ws,
                              size_t ws_size, hipStream_t stream) {
  const float* u       = (const float*)d_in[0];
  const float* W_in    = (const float*)d_in[1];
  const float* W_out   = (const float*)d_in[2];
  const float* A_log   = (const float*)d_in[3];
  const float* dt_bias = (const float*)d_in[4];
  (void)in_sizes; (void)n_in; (void)out_size;   // chunk_size hardcoded (=64)

  char* w = (char*)d_ws;
  size_t off = 0;
  auto carve = [&](size_t bytes) -> void* {
    void* p = w + off;
    off = (off + bytes + 255) & ~(size_t)255;
    return p;
  };
  bf16* U16    = (bf16*)carve((size_t)MROWS * DMODEL * 2);           // 32 MiB
  bf16* WIN16  = (bf16*)carve((size_t)KIN * DMODEL * 2);             // 17 MiB
  bf16* WOUT16 = (bf16*)carve((size_t)DMODEL * INNER * 2);           // 16 MiB
  float* XBCDT = (float*)carve((size_t)MROWS * KIN * 4);             // 138 MiB
  bf16* XDTT   = (bf16*)carve((size_t)MROWS * INNER * 2);            // 64 MiB
  bf16* BC     = (bf16*)carve((size_t)MROWS * DSTATE * 2);
  bf16* CC     = (bf16*)carve((size_t)MROWS * DSTATE * 2);
  bf16* BT     = (bf16*)carve((size_t)MROWS * DSTATE * 2);
  float* DTA   = (float*)carve((size_t)MROWS * NHEADS * 4);
  float* ACS   = (float*)carve((size_t)MROWS * NHEADS * 4);
  float* AEND  = (float*)carve((size_t)BATCH * NHEADS * NCHUNK * 4);
  bf16* GBUF   = (bf16*)carve((size_t)BATCH * NCHUNK * CHUNK * CHUNK * 2);
  bf16* ST     = (bf16*)carve((size_t)BATCH * NCHUNK * NHEADS * HEADDIM * DSTATE * 2); // 128 MiB
  bf16* YB     = (bf16*)carve((size_t)MROWS * INNER * 2);            // 64 MiB
  if (off > ws_size) return;  // workspace too small: nothing safe to do

  float* y_out  = (float*)d_out;                                   // (b,l,d)
  float* fs_out = y_out + (size_t)MROWS * DMODEL;                  // (b,h,p,n)

  // 0) pack to bf16
  pack_bf16_kernel<<<8192, 256, 0, stream>>>(u, U16, (long)MROWS * DMODEL);
  pack_bf16_kernel<<<8192, 256, 0, stream>>>(W_in, WIN16, (long)KIN * DMODEL);
  pack_bf16_kernel<<<8192, 256, 0, stream>>>(W_out, WOUT16, (long)DMODEL * INNER);

  // 1) xBCdt = u @ W_in^T   (M=8192, N=4416, K=2048)
  gemm_nt_kernel<<<dim3(KIN / 64, MROWS / 128), 256, 0, stream>>>(
      U16, WIN16, XBCDT, MROWS, KIN, DMODEL);

  // 2) split / softplus / scale / transposed packs
  prep_kernel<<<MROWS, 256, 0, stream>>>(XBCDT, A_log, dt_bias, XDTT, BC, CC,
                                         BT, DTA);
  // 3) per-chunk cumsum of A*dt
  cumsum_kernel<<<BATCH * NCHUNK, 64, 0, stream>>>(DTA, ACS, AEND);
  // 4) G = C B^T per chunk
  gmat_kernel<<<BATCH * NCHUNK, 128, 0, stream>>>(CC, BC, GBUF);
  // 5) local chunk states
  states_kernel<<<BATCH * NCHUNK * NHEADS, 128, 0, stream>>>(XDTT, BT, ACS, ST);
  // 6) chunk-level scan (in-place prefix) + final_state output
  scan_kernel<<<BATCH * NHEADS, 256, 0, stream>>>(AEND, ST, fs_out);
  // 7) Y = Y_diag + Y_off  (bf16, GEMM2-A layout)
  y_kernel<<<BATCH * NCHUNK * NHEADS, 128, 0, stream>>>(GBUF, CC, XDTT, ST,
                                                        ACS, YB);
  // 8) y = Y @ W_out^T  (M=8192, N=2048, K=4096)
  gemm_nt_kernel<<<dim3(DMODEL / 64, MROWS / 128), 256, 0, stream>>>(
      YB, WOUT16, y_out, MROWS, DMODEL, INNER);
}
